// LSTM_Model_Sequence_label_85091892068828
// MI455X (gfx1250) — matrix operand order repaired and checked by reading, not compiled, not run on hardware
//
#include <hip/hip_runtime.h>
#include <hip/hip_bf16.h>

// ---------------------------------------------------------------------------
// LSTM sequence labeler, CDNA5 (gfx1250) implementation.
// B=32, S=1024, D=512, 4D=2048, L=2, V=32000, MAXW=256.
//
// Strategy:
//  - bf16 data / f32 accumulate via v_wmma_f32_16x16x32_bf16 (wave32).
//  - Fold input projection into the recurrence: gates = [x_t | h_{t-1}] @ Wcat^T
//    with Wcat = [Wih | Whh] (bf16, K=1024), bias = b_ih + b_hh.
//  - Persistent recurrent kernel: 16 blocks x 128 threads (4 waves). Each wave
//    owns one (m_tile in 0..1, d_tile in 0..31) 16x16 patch of the hidden
//    state for ALL 1024 steps; cell state c stays in VGPRs the whole time.
//    h is double-buffered in global (32KB, L2-hot); [x_t|h] is staged in LDS.
//    Grid-wide barrier between steps via agent-scope atomics (16 blocks are
//    trivially co-resident on MI455X).
//  - Deterministic ragged mean-pool via binary search on sorted word ids,
//    fused with the 512->2 head + per-word NLL; fixed-order loss reduction.
// ---------------------------------------------------------------------------

typedef __bf16 bf16_t;
typedef __attribute__((ext_vector_type(16))) __bf16        v16bf;
typedef __attribute__((ext_vector_type(8)))  float         v8f;
typedef __attribute__((ext_vector_type(4)))  unsigned int  v4u;

#define NB   32      // batch
#define NS   1024    // sequence
#define ND   512     // hidden
#define NG   2048    // 4*ND
#define NK   1024    // folded K = ND(x) + ND(h)
#define NW   256     // MAXW
#define NWORDS (NB*NW)   // 8192
#define NBLK_LSTM 16

__device__ __forceinline__ float sigmoid_f(float x) { return 1.0f / (1.0f + __expf(-x)); }
__device__ __forceinline__ float tanh_f(float x)    { return 2.0f / (1.0f + __expf(-2.0f * x)) - 1.0f; }

// ---------------------------------------------------------------------------
// Grid-wide barrier (release on arrive, acquire on depart).
// ---------------------------------------------------------------------------
__device__ __forceinline__ void grid_barrier(int* cnt, int* gen, int nblk) {
    __threadfence();            // publish this thread's stores (agent scope)
    __syncthreads();
    if (threadIdx.x == 0) {
        int g = __hip_atomic_load(gen, __ATOMIC_RELAXED, __HIP_MEMORY_SCOPE_AGENT);
        if (__hip_atomic_fetch_add(cnt, 1, __ATOMIC_ACQ_REL, __HIP_MEMORY_SCOPE_AGENT) == nblk - 1) {
            __hip_atomic_store(cnt, 0, __ATOMIC_RELAXED, __HIP_MEMORY_SCOPE_AGENT);
            __hip_atomic_fetch_add(gen, 1, __ATOMIC_RELEASE, __HIP_MEMORY_SCOPE_AGENT);
        } else {
            while (__hip_atomic_load(gen, __ATOMIC_ACQUIRE, __HIP_MEMORY_SCOPE_AGENT) == g) {
                __builtin_amdgcn_s_sleep(2);
            }
        }
    }
    __syncthreads();
    __threadfence();            // invalidate near caches before reading peers' h
}

// ---------------------------------------------------------------------------
// Embedding gather: X0[b,s,:] = bf16(emb[x[b,s],:])
// ---------------------------------------------------------------------------
__global__ __launch_bounds__(128) void embed_kernel(const int* __restrict__ x,
                                                    const float* __restrict__ emb,
                                                    bf16_t* __restrict__ X0) {
    int tok = blockIdx.x;                      // 0 .. B*S-1
    int v = x[tok];
    const float* src = emb + (size_t)v * ND;
    bf16_t* dst = X0 + (size_t)tok * ND;
    for (int c = threadIdx.x; c < ND; c += 128)
        dst[c] = (bf16_t)src[c];
}

// ---------------------------------------------------------------------------
// Weight prep: Wcat[l][n][0:512]=Wih[l][n][:], [512:1024]=Whh[l][n][:] (bf16),
//              bias[l][n] = b_ih + b_hh.
// ---------------------------------------------------------------------------
__global__ __launch_bounds__(128) void prep_kernel(const float* __restrict__ W_ih,
                                                   const float* __restrict__ W_hh,
                                                   const float* __restrict__ b_ih,
                                                   const float* __restrict__ b_hh,
                                                   bf16_t* __restrict__ Wcat,
                                                   float* __restrict__ bias) {
    int row = blockIdx.x;                      // 0 .. L*NG-1
    int l = row >> 11, n = row & (NG - 1);
    const float* ih = W_ih + ((size_t)l * NG + n) * ND;
    const float* hh = W_hh + ((size_t)l * NG + n) * ND;
    bf16_t* dst = Wcat + (size_t)row * NK;
    for (int k = threadIdx.x; k < ND; k += 128) {
        dst[k]      = (bf16_t)ih[k];
        dst[ND + k] = (bf16_t)hh[k];
    }
    if (threadIdx.x == 0)
        bias[row] = b_ih[l * NG + n] + b_hh[l * NG + n];
}

// ---------------------------------------------------------------------------
// Zero h double-buffer + sync counters.
// ---------------------------------------------------------------------------
__global__ void zero_kernel(unsigned int* __restrict__ hbuf_u32, int* __restrict__ sync) {
    int i = blockIdx.x * blockDim.x + threadIdx.x;
    for (int q = i; q < (2 * NB * ND) / 2; q += gridDim.x * blockDim.x)  // 16384 u32
        hbuf_u32[q] = 0u;
    if (i < 2) sync[i] = 0;
}

// ---------------------------------------------------------------------------
// Persistent recurrent LSTM layer.
//   Xin  : [B,S,D] bf16 (layer input)
//   Wcat : [NG,NK] bf16  (this layer)    bias: [NG] f32
//   Hbuf : [2][B][D] bf16 double buffer  (buf0 pre-zeroed = h0)
//   HSout: [B,S,D] bf16 (layer output)
// wave (blockIdx*4+waveId) -> m_tile = wg>>5 (const per block), d_tile = wg&31.
// ---------------------------------------------------------------------------
__global__ __launch_bounds__(128, 1) void lstm_layer_kernel(
        const bf16_t* __restrict__ Xin,
        const bf16_t* __restrict__ Wcat,
        const float*  __restrict__ bias,
        bf16_t* __restrict__ Hbuf,
        bf16_t* __restrict__ HSout,
        int* sync_cnt, int* sync_gen) {
    __shared__ __align__(16) unsigned short sA[16 * NK];   // [16 rows][1024 k] bf16 = 32KB

    const int tid    = threadIdx.x;
    const int lane   = tid & 31;
    const int wave   = tid >> 5;
    const int wg     = blockIdx.x * 4 + wave;              // 0..63
    const int m_tile = blockIdx.x >> 3;                    // 0..1 (uniform per block)
    const int d_tile = wg & 31;                            // 0..31
    const int half   = lane >> 4;                          // 0..1
    const int nl     = lane & 15;                          // col / A-row within tile

    // Per-lane base offsets into Wcat for the 4 gate rows (i,f,g,o).
    size_t brow[4];
    v8f bias_v[4];
#pragma unroll
    for (int gi = 0; gi < 4; ++gi) {
        int n = gi * ND + d_tile * 16 + nl;
        brow[gi] = (size_t)n * NK;
        float bv = bias[n];
#pragma unroll
        for (int r = 0; r < 8; ++r) bias_v[gi][r] = bv;
    }

    v8f cstate = {};   // cell state lives in VGPRs for the whole sequence

#pragma unroll 1
    for (int t = 0; t < NS; ++t) {
        grid_barrier(sync_cnt, sync_gen, NBLK_LSTM);

        // ---- stage A = [x_t | h_{t-1}] rows (this block's 16 batch rows) into LDS
        const bf16_t* Hprev = Hbuf + (size_t)(t & 1) * (NB * ND);
        for (int q = tid; q < 2048; q += 128) {            // 2048 x 16B = 32KB
            int row = q >> 7;                              // 0..15
            int c0  = (q & 127) << 3;                      // 0..1016 step 8
            int rg  = m_tile * 16 + row;                   // batch row
            const bf16_t* src = (c0 < ND)
                ? (Xin + ((size_t)rg * NS + t) * ND + c0)
                : (Hprev + (size_t)rg * ND + (c0 - ND));
            *(v4u*)&sA[row * NK + c0] = *(const v4u*)src;
        }
        __syncthreads();

        // ---- gates = bias + A @ Wcat^T  (f32 accumulate, bf16 WMMA)
        v8f acc[4] = { bias_v[0], bias_v[1], bias_v[2], bias_v[3] };
#pragma unroll 4
        for (int kk = 0; kk < NK; kk += 32) {
            // A fragment (16x32 bf16): lane half h: elems 0..15 <- k=kk+16h..+15
            // (contiguous, matching B)
            v16bf a;
            {
                const unsigned short* pa = &sA[nl * NK + kk + 16 * half];
                v4u lo = *(const v4u*)pa;
                v4u hi = *(const v4u*)(pa + 8);
                __builtin_memcpy(&a, &lo, 16);
                __builtin_memcpy((char*)&a + 16, &hi, 16);
            }
#pragma unroll
            for (int gi = 0; gi < 4; ++gi) {
                // B fragment (32x16 bf16): lane = col nl, K run = kk+16*half..+15
                const bf16_t* pb = Wcat + brow[gi] + kk + 16 * half;
                v4u lo = *(const v4u*)pb;
                v4u hi = *(const v4u*)(pb + 8);
                v16bf b;
                __builtin_memcpy(&b, &lo, 16);
                __builtin_memcpy((char*)&b + 16, &hi, 16);
                acc[gi] = __builtin_amdgcn_wmma_f32_16x16x32_bf16(
                    /*neg_a=*/false, a, /*neg_b=*/false, b,
                    /*c_mod=*/(short)0, acc[gi], /*reuse_a=*/false, /*reuse_b=*/false);
            }
        }

        // ---- LSTM cell (torch gate order i,f,g,o); write h for next step + output
        bf16_t* Hnext = Hbuf + (size_t)((t + 1) & 1) * (NB * ND);
        const int d = d_tile * 16 + nl;
#pragma unroll
        for (int r = 0; r < 8; ++r) {
            float iv = sigmoid_f(acc[0][r]);
            float fv = sigmoid_f(acc[1][r]);
            float gv = tanh_f(acc[2][r]);
            float ov = sigmoid_f(acc[3][r]);
            float c  = fv * cstate[r] + iv * gv;
            cstate[r] = c;
            float h  = ov * tanh_f(c);
            int rowg = m_tile * 16 + r + 8 * half;         // C/D layout: M = r + 8*half
            bf16_t hb = (bf16_t)h;
            Hnext[(size_t)rowg * ND + d] = hb;
            HSout[((size_t)rowg * NS + t) * ND + d] = hb;
        }
        // stores published by the fence at the top of next grid_barrier
    }
}

// ---------------------------------------------------------------------------
// Ragged mean pool + span head + per-word NLL (deterministic: binary search
// on sorted token_word_id, fixed-order reductions). One block per word.
// ---------------------------------------------------------------------------
__global__ __launch_bounds__(64) void pool_head_kernel(
        const bf16_t* __restrict__ HS,      // [B,S,D] final layer
        const int* __restrict__ twid,       // [B,S] sorted per row
        const int* __restrict__ labels,     // [B,MAXW]
        const float* __restrict__ Wout,     // [2,D]
        const float* __restrict__ bout,     // [2]
        float* __restrict__ logits,         // [B*MAXW,2] -> d_out
        float* __restrict__ nll) {          // [B*MAXW]
    int word = blockIdx.x;                  // 0..8191
    int b = word >> 8;
    int w = word & (NW - 1);

    __shared__ int s_lo, s_hi;
    __shared__ float red0[64], red1[64];
    if (threadIdx.x == 0) {
        const int* row = twid + (size_t)b * NS;
        int lo = 0, hi = NS;
        while (lo < hi) { int mid = (lo + hi) >> 1; if (row[mid] <  w) lo = mid + 1; else hi = mid; }
        s_lo = lo;
        int lo2 = lo, hi2 = NS;
        while (lo2 < hi2) { int mid = (lo2 + hi2) >> 1; if (row[mid] <= w) lo2 = mid + 1; else hi2 = mid; }
        s_hi = lo2;
    }
    __syncthreads();
    const int lo = s_lo, hi = s_hi;
    const float cnt = (float)(hi - lo);

    float p0 = 0.f, p1 = 0.f;
#pragma unroll
    for (int j = 0; j < 8; ++j) {
        int col = threadIdx.x + 64 * j;
        float sum = 0.f;
        for (int s = lo; s < hi; ++s)
            sum += (float)HS[((size_t)b * NS + s) * ND + col];
        float pooled = (cnt > 0.f) ? (sum / cnt) : 0.f;
        p0 += pooled * Wout[col];
        p1 += pooled * Wout[ND + col];
    }
    red0[threadIdx.x] = p0; red1[threadIdx.x] = p1;
    __syncthreads();
    for (int off = 32; off > 0; off >>= 1) {
        if (threadIdx.x < off) {
            red0[threadIdx.x] += red0[threadIdx.x + off];
            red1[threadIdx.x] += red1[threadIdx.x + off];
        }
        __syncthreads();
    }
    if (threadIdx.x == 0) {
        float l0 = red0[0] + bout[0];
        float l1 = red1[0] + bout[1];
        logits[word * 2 + 0] = l0;
        logits[word * 2 + 1] = l1;
        float m = fmaxf(l0, l1);
        float lse = m + __logf(__expf(l0 - m) + __expf(l1 - m));
        int lab = labels[word];
        nll[word] = lse - (lab ? l1 : l0);
    }
}

// ---------------------------------------------------------------------------
// loss = mean(nll) over 8192 words, fixed-order tree reduction.
// ---------------------------------------------------------------------------
__global__ __launch_bounds__(256) void loss_kernel(const float* __restrict__ nll,
                                                   float* __restrict__ out_loss) {
    __shared__ float red[256];
    float s = 0.f;
    for (int i = threadIdx.x; i < NWORDS; i += 256) s += nll[i];
    red[threadIdx.x] = s;
    __syncthreads();
    for (int off = 128; off > 0; off >>= 1) {
        if (threadIdx.x < off) red[threadIdx.x] += red[threadIdx.x + off];
        __syncthreads();
    }
    if (threadIdx.x == 0) *out_loss = red[0] / (float)NWORDS;
}

// ---------------------------------------------------------------------------
// Workspace layout (all offsets 256B aligned), ~104 MB total.
// ---------------------------------------------------------------------------
#define OFF_X0    ((size_t)0)                                // 32MB  bf16 [B,S,D]
#define OFF_HS1   ((size_t)33554432)                         // 32MB  bf16
#define OFF_HS2   ((size_t)67108864)                         // 32MB  bf16
#define OFF_WCAT  ((size_t)100663296)                        // 8MB   bf16 [L,NG,NK]
#define OFF_BIAS  ((size_t)109051904)                        // 16KB  f32  [L,NG]
#define OFF_HBUF  ((size_t)109068288)                        // 64KB  bf16 [2,B,D]
#define OFF_NLL   ((size_t)109133824)                        // 32KB  f32
#define OFF_SYNC  ((size_t)109166592)                        // counters

extern "C" void kernel_launch(void* const* d_in, const int* in_sizes, int n_in,
                              void* d_out, int out_size, void* d_ws, size_t ws_size,
                              hipStream_t stream) {
    (void)in_sizes; (void)n_in; (void)out_size; (void)ws_size;
    const int*   x      = (const int*)  d_in[0];
    const int*   labels = (const int*)  d_in[1];
    const int*   twid   = (const int*)  d_in[2];
    const float* emb    = (const float*)d_in[3];
    const float* W_ih   = (const float*)d_in[4];
    const float* W_hh   = (const float*)d_in[5];
    const float* b_ih   = (const float*)d_in[6];
    const float* b_hh   = (const float*)d_in[7];
    const float* W_out  = (const float*)d_in[8];
    const float* b_out  = (const float*)d_in[9];

    char* ws = (char*)d_ws;
    bf16_t* X0   = (bf16_t*)(ws + OFF_X0);
    bf16_t* HS1  = (bf16_t*)(ws + OFF_HS1);
    bf16_t* HS2  = (bf16_t*)(ws + OFF_HS2);
    bf16_t* Wcat = (bf16_t*)(ws + OFF_WCAT);
    float*  bias = (float*) (ws + OFF_BIAS);
    bf16_t* Hbuf = (bf16_t*)(ws + OFF_HBUF);
    float*  nll  = (float*) (ws + OFF_NLL);
    int*    sync = (int*)   (ws + OFF_SYNC);
    float*  out  = (float*)d_out;

    // 1. embedding -> bf16
    embed_kernel<<<NB * NS, 128, 0, stream>>>(x, emb, X0);
    // 2. weights -> [Wih|Whh] bf16 + fused bias
    prep_kernel<<<2 * NG, 128, 0, stream>>>(W_ih, W_hh, b_ih, b_hh, Wcat, bias);
    // 3. layer 1
    zero_kernel<<<16, 256, 0, stream>>>((unsigned int*)Hbuf, sync);
    lstm_layer_kernel<<<NBLK_LSTM, 128, 0, stream>>>(X0, Wcat, bias, Hbuf, HS1,
                                                     sync + 0, sync + 1);
    // 4. layer 2
    zero_kernel<<<16, 256, 0, stream>>>((unsigned int*)Hbuf, sync);
    lstm_layer_kernel<<<NBLK_LSTM, 128, 0, stream>>>(HS1, Wcat + (size_t)NG * NK,
                                                     bias + NG, Hbuf, HS2,
                                                     sync + 0, sync + 1);
    // 5. pool + head + per-word NLL (logits straight into d_out)
    pool_head_kernel<<<NWORDS, 64, 0, stream>>>(HS2, twid, labels, W_out, b_out, out, nll);
    // 6. loss -> d_out[16384]
    loss_kernel<<<1, 256, 0, stream>>>(nll, out + NWORDS * 2);
}
